// _DiffCrossAttnDirection_42116449305217
// MI455X (gfx1250) — matrix operand order repaired and checked by reading, not compiled, not run on hardware
//
#include <hip/hip_runtime.h>
#include <hip/hip_bf16.h>

#define DEVI __device__ __forceinline__

typedef __bf16 bf16_t;
typedef __attribute__((ext_vector_type(8)))  __bf16 v8bf;
typedef __attribute__((ext_vector_type(16))) __bf16 v16bf;
typedef __attribute__((ext_vector_type(8)))  float  v8f;

static constexpr int B_ = 4, S_ = 1024, D_ = 1024, H_ = 16, HD_ = 64, DH2_ = 128, E_ = 2048;
static constexpr float EPSV = 1e-5f;

// ---------------- WMMA fragment helpers (CDNA5 16x16x32 bf16) ----------------
// A layout: lane<16 -> row=lane,    K {k0..k0+7, k0+16..k0+23}
//           lane>=16 -> row=lane-16, K {k0+8..k0+15, k0+24..k0+31}
DEVI v16bf frag_a(const bf16_t* __restrict__ base, int ld, int row0, int k0, int lane) {
  int r  = row0 + (lane & 15);
  int kk = k0 + ((lane >> 4) << 3);
  const bf16_t* p = base + (size_t)r * ld + kk;
  v8bf lo = *(const v8bf*)p;
  v8bf hi = *(const v8bf*)(p + 16);
  v16bf o;
#pragma unroll
  for (int i = 0; i < 8; ++i) { o[i] = lo[i]; o[8 + i] = hi[i]; }
  return o;
}

// B layout: lane<16 -> col=lane,    K {k0..k0+7, k0+16..k0+23}
//           lane>=16 -> col=lane-16, K {k0+8..k0+15, k0+24..k0+31}
// Source is B^T row-major (N x K, K contiguous) -> two 16-byte chunks per lane.
DEVI v16bf frag_b(const bf16_t* __restrict__ base, int ld, int n0, int k0, int lane) {
  int nn = n0 + (lane & 15);
  int kk = k0 + ((lane >> 4) << 3);
  const bf16_t* p = base + (size_t)nn * ld + kk;
  v8bf lo = *(const v8bf*)p;
  v8bf hi = *(const v8bf*)(p + 16);
  v16bf o;
#pragma unroll
  for (int i = 0; i < 8; ++i) { o[i] = lo[i]; o[8 + i] = hi[i]; }
  return o;
}

DEVI v8f wmma_bf16(v16bf a, v16bf b, v8f c) {
  return __builtin_amdgcn_wmma_f32_16x16x32_bf16(false, a, false, b, (short)0, c, false, false);
}

DEVI v8f vzero8() { v8f z;
#pragma unroll
  for (int i = 0; i < 8; ++i) z[i] = 0.f; return z; }

// ---- DPP16 butterfly reductions within each 16-lane half (pure VALU, no DS) ----
template <int CTRL>
DEVI float dppf(float v) {
  return __int_as_float(
      __builtin_amdgcn_update_dpp(0, __float_as_int(v), CTRL, 0xF, 0xF, true));
}
DEVI float hmax16(float v) {
  v = fmaxf(v, dppf<0xB1>(v));   // quad_perm(1,0,3,2)  : xor 1
  v = fmaxf(v, dppf<0x4E>(v));   // quad_perm(2,3,0,1)  : xor 2
  v = fmaxf(v, dppf<0x141>(v));  // row_half_mirror     : cross quads in 8
  v = fmaxf(v, dppf<0x140>(v));  // row_mirror          : cross 8s in 16
  return v;
}
DEVI float hsum16(float v) {
  v += dppf<0xB1>(v);
  v += dppf<0x4E>(v);
  v += dppf<0x141>(v);
  v += dppf<0x140>(v);
  return v;
}
DEVI float wsum32(float v) {
#pragma unroll
  for (int off = 16; off; off >>= 1) v += __shfl_xor(v, off, 32);
  return v;
}

// ---------------- utility kernels ----------------
__global__ void cvt_bf16(const float* __restrict__ in, bf16_t* __restrict__ out, int n) {
  for (int i = blockIdx.x * blockDim.x + threadIdx.x; i < n; i += gridDim.x * blockDim.x)
    out[i] = (bf16_t)in[i];
}

// LayerNorm over 1024-wide rows, write bf16
__global__ __launch_bounds__(256) void ln_rows(const float* __restrict__ x,
                                               const float* __restrict__ g,
                                               const float* __restrict__ bb,
                                               bf16_t* __restrict__ y) {
  const int row = blockIdx.x;
  const int tid = threadIdx.x;
  const int lane = tid & 31, wave = tid >> 5;
  const float* xr = x + (size_t)row * D_;
  float v[4]; float s = 0.f;
#pragma unroll
  for (int i = 0; i < 4; ++i) { v[i] = xr[tid + i * 256]; s += v[i]; }
  s = wsum32(s);
  __shared__ float rA[8], rB[8];
  if (lane == 0) rA[wave] = s;
  __syncthreads();
  float tot = 0.f;
#pragma unroll
  for (int w = 0; w < 8; ++w) tot += rA[w];
  float mean = tot * (1.f / 1024.f);
  float q = 0.f;
#pragma unroll
  for (int i = 0; i < 4; ++i) { float d = v[i] - mean; q += d * d; }
  q = wsum32(q);
  if (lane == 0) rB[wave] = q;
  __syncthreads();
  float qt = 0.f;
#pragma unroll
  for (int w = 0; w < 8; ++w) qt += rB[w];
  float rstd = rsqrtf(qt * (1.f / 1024.f) + EPSV);
#pragma unroll
  for (int i = 0; i < 4; ++i) {
    int idx = tid + i * 256;
    y[(size_t)row * D_ + idx] = (bf16_t)((v[i] - mean) * rstd * g[idx] + bb[idx]);
  }
}

// lambda = exp(q1.k1) - exp(q2.k2) + lambda_init   (one wave)
__global__ void lambda_kernel(const float* __restrict__ q1, const float* __restrict__ k1,
                              const float* __restrict__ q2, const float* __restrict__ k2,
                              const float* __restrict__ li, const int* __restrict__ dis,
                              float* __restrict__ lam) {
  int l = threadIdx.x;
  float d1 = q1[l] * k1[l] + q1[l + 32] * k1[l + 32];
  float d2 = q2[l] * k2[l] + q2[l + 32] * k2[l + 32];
  d1 = wsum32(d1); d2 = wsum32(d2);
  if (l == 0) lam[0] = dis[0] ? 0.f : (__expf(d1) - __expf(d2) + li[0]);
}

// ---------------- generic TN WMMA GEMM: C(MxN) = A(MxK) * Bm(NxK)^T ----------------
// Wave tile 32x64 (2 A-frags share 4 B-frags -> 8 WMMA per 12 b128 loads).
// Block tile 256x64 (8 waves). cscale folded into the store (exact for powers of 2).
// MODE 1: bf16 scatter -> (B,H,S,128)  (q/k projection)
// MODE 2: bf16 scatter -> (B,H,128,S)  (v projection, transposed)
// MODE 3: f32 row-major + residual add (final projection)
template <int MODE>
__global__ __launch_bounds__(256) void gemm_tn(const bf16_t* __restrict__ A,
                                               const bf16_t* __restrict__ Bm,
                                               void* __restrict__ dstv,
                                               const float* __restrict__ resid,
                                               int M, int N, int K, float cscale) {
  const int lane = threadIdx.x & 31;
  const int wave = threadIdx.x >> 5;
  const int m0 = blockIdx.y * 256 + wave * 32;
  const int n0 = blockIdx.x * 64;
  v8f acc[2][4];
#pragma unroll
  for (int h = 0; h < 2; ++h)
#pragma unroll
    for (int t = 0; t < 4; ++t) acc[h][t] = vzero8();
  for (int k0 = 0; k0 < K; k0 += 32) {
    v16bf a0 = frag_a(A, K, m0, k0, lane);
    v16bf a1 = frag_a(A, K, m0 + 16, k0, lane);
#pragma unroll
    for (int t = 0; t < 4; ++t) {
      v16bf b = frag_b(Bm, K, n0 + t * 16, k0, lane);
      acc[0][t] = wmma_bf16(a0, b, acc[0][t]);
      acc[1][t] = wmma_bf16(a1, b, acc[1][t]);
    }
  }
  const int rofs = (lane >> 4) << 3;
  const int c = lane & 15;
#pragma unroll
  for (int h = 0; h < 2; ++h) {
#pragma unroll
    for (int t = 0; t < 4; ++t) {
#pragma unroll
      for (int g = 0; g < 8; ++g) {
        float val = acc[h][t][g] * cscale;
        int m = m0 + h * 16 + g + rofs;
        int n = n0 + t * 16 + c;
        if constexpr (MODE == 1 || MODE == 2) {
          int bbi = m >> 10, s = m & (S_ - 1);
          int hh = n >> 7, dd = n & 127;
          size_t idx = (MODE == 1)
              ? ((((size_t)bbi * H_ + hh) * S_ + s) * DH2_ + dd)
              : ((((size_t)bbi * H_ + hh) * DH2_ + dd) * (size_t)S_ + s);
          ((bf16_t*)dstv)[idx] = (bf16_t)val;
        } else {
          size_t idx = (size_t)m * N + n;
          ((float*)dstv)[idx] = resid[idx] + val;
        }
      }
    }
  }
}

// ---------------- flash-style differential attention (one softmax term) ----------------
// grid: (S/128, B*H); 8 waves, each owns 16 q rows. doff selects q1/k1 (0) or q2/k2 (64).
// Scores arrive pre-scaled (0.125 folded into q projection).
// pass 0: out = attn@v ; pass 1: out -= lam * attn@v
__global__ __launch_bounds__(256) void attn_kernel(const bf16_t* __restrict__ qp,
                                                   const bf16_t* __restrict__ kp,
                                                   const bf16_t* __restrict__ vt,
                                                   float* __restrict__ out,
                                                   const float* __restrict__ lamp,
                                                   int doff, int pass) {
  __shared__ __align__(16) bf16_t pbuf[8 * 16 * 32];
  const int lane = threadIdx.x & 31;
  const int wave = threadIdx.x >> 5;
  const int bh = blockIdx.y;
  const int s0 = blockIdx.x * 128 + wave * 16;
  const bf16_t* qb = qp + (size_t)bh * S_ * DH2_;
  const bf16_t* kb = kp + (size_t)bh * S_ * DH2_;
  const bf16_t* vb = vt + (size_t)bh * DH2_ * S_;

  v16bf qf0 = frag_a(qb, DH2_, s0, doff, lane);
  v16bf qf1 = frag_a(qb, DH2_, s0, doff + 32, lane);

  v8f o[8];
  float mrow[8], lrow[8];
#pragma unroll
  for (int n = 0; n < 8; ++n) o[n] = vzero8();
#pragma unroll
  for (int g = 0; g < 8; ++g) { mrow[g] = -1e30f; lrow[g] = 0.f; }

  bf16_t* pb = pbuf + wave * (16 * 32);
  const int rofs = (lane >> 4) << 3;
  const int c = lane & 15;

  for (int t0 = 0; t0 < S_; t0 += 32) {
    // scores: q(16x64) @ k^T(64x32) -> two 16x16 f32 tiles (already scaled)
    v8f c0 = vzero8(), c1 = vzero8();
    {
      v16bf kf0 = frag_b(kb, DH2_, t0, doff, lane);
      v16bf kf1 = frag_b(kb, DH2_, t0, doff + 32, lane);
      c0 = wmma_bf16(qf0, kf0, c0);
      c0 = wmma_bf16(qf1, kf1, c0);
      v16bf kf2 = frag_b(kb, DH2_, t0 + 16, doff, lane);
      v16bf kf3 = frag_b(kb, DH2_, t0 + 16, doff + 32, lane);
      c1 = wmma_bf16(qf0, kf2, c1);
      c1 = wmma_bf16(qf1, kf3, c1);
    }
    // online softmax; row stats uniform across each 16-lane half, DPP butterflies
    float alpha[8];
#pragma unroll
    for (int g = 0; g < 8; ++g) {
      float a = c0[g], b2 = c1[g];
      float mx = hmax16(fmaxf(a, b2));
      float mn = fmaxf(mrow[g], mx);
      float al = __expf(mrow[g] - mn);
      mrow[g] = mn;
      a = __expf(a - mn); b2 = __expf(b2 - mn);
      c0[g] = a; c1[g] = b2;
      lrow[g] = lrow[g] * al + hsum16(a + b2);
      alpha[g] = al;
    }
#pragma unroll
    for (int n = 0; n < 8; ++n)
#pragma unroll
      for (int g = 0; g < 8; ++g) o[n][g] *= alpha[g];

    // P (C layout) -> LDS -> re-read as A fragment (wave-private, DS pipe is in-order)
#pragma unroll
    for (int g = 0; g < 8; ++g) {
      int r = g + rofs;
      pb[r * 32 + c] = (bf16_t)c0[g];
      pb[r * 32 + 16 + c] = (bf16_t)c1[g];
    }
    asm volatile("" ::: "memory");
    v16bf pf = frag_a(pb, 32, 0, 0, lane);
    asm volatile("" ::: "memory");

    // P(16x32) @ V(32x128) using transposed V (rows = d, t contiguous)
#pragma unroll
    for (int n = 0; n < 8; ++n) {
      v16bf vf = frag_b(vb, S_, n * 16, t0, lane);
      o[n] = wmma_bf16(pf, vf, o[n]);
    }
  }

  float lam = (pass == 1) ? lamp[0] : 0.f;
  float* ob = out + ((size_t)bh * S_ + s0) * DH2_;
#pragma unroll
  for (int n = 0; n < 8; ++n) {
#pragma unroll
    for (int g = 0; g < 8; ++g) {
      float val = o[n][g] / lrow[g];
      int r = g + rofs;
      size_t idx = (size_t)r * DH2_ + n * 16 + c;
      if (pass == 0) ob[idx] = val;
      else           ob[idx] = ob[idx] - lam * val;
    }
  }
}

// ---------------- head-wise LayerNorm over 128 dims, one wave per row ----------------
__global__ __launch_bounds__(256) void head_ln(const float* __restrict__ x,
                                               const float* __restrict__ g,
                                               const float* __restrict__ bb,
                                               const float* __restrict__ lamInit,
                                               bf16_t* __restrict__ y) {
  int lane = threadIdx.x & 31;
  int wave = threadIdx.x >> 5;
  int gid = blockIdx.x * 8 + wave;       // (b*H + h)*S + s
  int s = gid & (S_ - 1);
  int bh = gid >> 10;
  int h = bh & (H_ - 1);
  int bbi = bh >> 4;
  const float* xr = x + (size_t)gid * DH2_;
  float v[4]; float sum = 0.f;
#pragma unroll
  for (int i = 0; i < 4; ++i) { v[i] = xr[lane + i * 32]; sum += v[i]; }
  sum = wsum32(sum);
  float mean = sum * (1.f / 128.f);
  float q = 0.f;
#pragma unroll
  for (int i = 0; i < 4; ++i) { float d = v[i] - mean; q += d * d; }
  q = wsum32(q);
  float rstd = rsqrtf(q * (1.f / 128.f) + EPSV);
  float sc = 1.f - lamInit[0];
  size_t base = ((size_t)(bbi * S_ + s)) * E_ + (size_t)h * DH2_;
#pragma unroll
  for (int i = 0; i < 4; ++i) {
    int dd = lane + i * 32;
    y[base + dd] = (bf16_t)(((v[i] - mean) * rstd * g[dd] + bb[dd]) * sc);
  }
}

// ---------------- launch ----------------
extern "C" void kernel_launch(void* const* d_in, const int* in_sizes, int n_in,
                              void* d_out, int out_size, void* d_ws, size_t ws_size,
                              hipStream_t stream) {
  (void)in_sizes; (void)n_in; (void)out_size; (void)ws_size;
  const float* qtok   = (const float*)d_in[0];
  const float* kvtok  = (const float*)d_in[1];
  const float* lnqg   = (const float*)d_in[2];
  const float* lnqb   = (const float*)d_in[3];
  const float* lnkvg  = (const float*)d_in[4];
  const float* lnkvb  = (const float*)d_in[5];
  const float* Wq     = (const float*)d_in[6];
  const float* Wk     = (const float*)d_in[7];
  const float* Wv     = (const float*)d_in[8];
  const float* Wo     = (const float*)d_in[9];
  const float* lq1    = (const float*)d_in[10];
  const float* lk1    = (const float*)d_in[11];
  const float* lq2    = (const float*)d_in[12];
  const float* lk2    = (const float*)d_in[13];
  const float* hng    = (const float*)d_in[14];
  const float* hnb    = (const float*)d_in[15];
  const float* lamIni = (const float*)d_in[16];
  const int*   dis    = (const int*)d_in[17];

  char* ws = (char*)d_ws;
  bf16_t* qln     = (bf16_t*)(ws + 0);                   // 8 MB
  bf16_t* kvln    = (bf16_t*)(ws + (size_t)(8u  << 20)); // 8 MB
  bf16_t* attn_ln = (bf16_t*)(ws + 0);                   // alias over qln+kvln (16 MB)
  bf16_t* Wqb     = (bf16_t*)(ws + (size_t)(16u << 20));
  bf16_t* Wkb     = (bf16_t*)(ws + (size_t)(20u << 20));
  bf16_t* Wvb     = (bf16_t*)(ws + (size_t)(24u << 20));
  bf16_t* Wob     = (bf16_t*)(ws + (size_t)(28u << 20));
  bf16_t* qproj   = (bf16_t*)(ws + (size_t)(32u << 20)); // (B,H,S,128)
  bf16_t* kproj   = (bf16_t*)(ws + (size_t)(48u << 20)); // (B,H,S,128)
  bf16_t* vtproj  = (bf16_t*)(ws + (size_t)(64u << 20)); // (B,H,128,S)
  float*  acc     = (float*) (ws + (size_t)(80u << 20)); // (B,H,S,128) f32, 32 MB
  float*  lam     = (float*) (ws + (size_t)(112u << 20));

  const int nW = 2 * D_ * D_;  // 2M elements each
  cvt_bf16<<<2048, 256, 0, stream>>>(Wq, Wqb, nW);
  cvt_bf16<<<2048, 256, 0, stream>>>(Wk, Wkb, nW);
  cvt_bf16<<<2048, 256, 0, stream>>>(Wv, Wvb, nW);
  cvt_bf16<<<2048, 256, 0, stream>>>(Wo, Wob, nW);

  ln_rows<<<B_ * S_, 256, 0, stream>>>(qtok, lnqg, lnqb, qln);
  ln_rows<<<B_ * S_, 256, 0, stream>>>(kvtok, lnkvg, lnkvb, kvln);
  lambda_kernel<<<1, 32, 0, stream>>>(lq1, lk1, lq2, lk2, lamIni, dis, lam);

  // block tile 256x64
  dim3 gp(E_ / 64, (B_ * S_) / 256);  // (32, 16)
  gemm_tn<1><<<gp, 256, 0, stream>>>(qln,  Wqb, (void*)qproj,  nullptr, B_ * S_, E_, D_, 0.125f);
  gemm_tn<1><<<gp, 256, 0, stream>>>(kvln, Wkb, (void*)kproj,  nullptr, B_ * S_, E_, D_, 1.0f);
  gemm_tn<2><<<gp, 256, 0, stream>>>(kvln, Wvb, (void*)vtproj, nullptr, B_ * S_, E_, D_, 1.0f);

  dim3 ga(S_ / 128, B_ * H_);  // (8, 64)
  attn_kernel<<<ga, 256, 0, stream>>>(qproj, kproj, vtproj, acc, lam, 0, 0);
  attn_kernel<<<ga, 256, 0, stream>>>(qproj, kproj, vtproj, acc, lam, HD_, 1);

  head_ln<<<(B_ * H_ * S_) / 8, 256, 0, stream>>>(acc, hng, hnb, lamIni, attn_ln);

  dim3 go(D_ / 64, (B_ * S_) / 256);  // (16, 16)
  gemm_tn<3><<<go, 256, 0, stream>>>(attn_ln, Wob, d_out, qtok, B_ * S_, D_, E_, 1.0f);
}
